// FGNet3D_35270271435549
// MI455X (gfx1250) — matrix operand order repaired and checked by reading, not compiled, not run on hardware
//
#include <hip/hip_runtime.h>

typedef __attribute__((ext_vector_type(16))) _Float16      v16h;
typedef __attribute__((ext_vector_type(8)))  float         v8f;
typedef __attribute__((ext_vector_type(4)))  unsigned int  v4u;
typedef __attribute__((ext_vector_type(8)))  int           v8i;
typedef __attribute__((ext_vector_type(4)))  int           v4i;

#if __has_builtin(__builtin_amdgcn_tensor_load_to_lds)
#define HAVE_TDM 1
#else
#define HAVE_TDM 0
#endif

#define RFL(v) __builtin_amdgcn_readfirstlane((int)(v))

// ---------------------------------------------------------------------------
// TDM helper: issue a 2D tile load (tile_dim0 x tile_dim1 elements of f16)
// from packed weights into LDS.  D# built per CDNA5 ISA.
// ---------------------------------------------------------------------------
#if HAVE_TDM
__device__ __forceinline__ void tdm_load_2d_f16(unsigned ldsOff,
                                                unsigned long gaddr,
                                                int td0, int td1,
                                                int tile0, int tile1,
                                                int stride0)
{
    v4u g0;
    g0[0] = 1u;                                            // count=1, user D#
    g0[1] = (unsigned)RFL(ldsOff);                         // LDS byte address
    g0[2] = (unsigned)RFL((unsigned)gaddr);                // global addr lo
    g0[3] = (unsigned)((RFL((unsigned)(gaddr >> 32)) & 0x01ffffff) | (2u << 30)); // hi + type=2
    v8i g1;
    g1[0] = 0x00010000;                                    // data_size=1 (2 bytes)
    g1[1] = (RFL(td0) & 0xffff) << 16;                     // tensor_dim0[15:0]
    g1[2] = ((unsigned)RFL(td0) >> 16) | ((RFL(td1) & 0xffff) << 16);
    g1[3] = ((unsigned)RFL(td1) >> 16) | (RFL(tile0) << 16);  // tile_dim0
    g1[4] = RFL(tile1);                                    // tile_dim1 (tile_dim2=0)
    g1[5] = RFL(stride0);                                  // tensor_dim0_stride lo32
    g1[6] = 0;
    g1[7] = 0;
    v4i z4 = {0, 0, 0, 0};
#if __clang_major__ >= 23
    v8i z8 = {0, 0, 0, 0, 0, 0, 0, 0};
    __builtin_amdgcn_tensor_load_to_lds(g0, g1, z4, z4, z8, 0);
#else
    __builtin_amdgcn_tensor_load_to_lds(g0, g1, z4, z4, 0);
#endif
}
#endif

// Stage one 32(k') x 16(n) f16 B tile into LDS (row-major [n][32]).
__device__ __forceinline__ void stage_b_tile(const _Float16* __restrict__ wP,
                                             int nBase, int k0p, int Kp, int Np,
                                             _Float16* shDst)
{
#if HAVE_TDM
    if (threadIdx.x < 32) {
        unsigned ldsOff = (unsigned)(unsigned long long)shDst;
        unsigned long ga = (unsigned long)(const void*)(wP + (long)nBase * Kp + k0p);
        tdm_load_2d_f16(ldsOff, ga, Kp, Np, 32, 16, Kp);
    }
#else
    for (int i = threadIdx.x; i < 512; i += 256) {
        int nn = i >> 5, kk = i & 31;
        shDst[i] = wP[(long)(nBase + nn) * Kp + (k0p + kk)];
    }
#endif
}

__device__ __forceinline__ void stage_wait()
{
#if HAVE_TDM
    __builtin_amdgcn_s_wait_tensorcnt(0);   // no-op for waves with TENSORcnt==0
#endif
}

// ---------------------------------------------------------------------------
// Weight repack: (Cout,Cin,KT) f32 -> tap-major padded f16  wP[n][k'],
// k' = tap*Cp + cin, Kp = ceil(KT*Cp/32)*32, Np = ceil(Cout/16)*16.
// ---------------------------------------------------------------------------
__global__ __launch_bounds__(256)
void k_pack_w(const float* __restrict__ w, _Float16* __restrict__ wP,
              int Cin, int Cout, int KT, int Cp, int Kp, int Np, int lcp)
{
    long i = (long)blockIdx.x * 256 + threadIdx.x;
    long tot = (long)Np * Kp;
    if (i >= tot) return;
    int n  = (int)(i / Kp);
    int kp = (int)(i - (long)n * Kp);
    int tap = kp >> lcp;
    int cin = kp & (Cp - 1);
    float v = 0.f;
    if (n < Cout && tap < KT && cin < Cin)
        v = w[((long)n * Cin + cin) * KT + tap];
    wP[i] = (_Float16)v;
}

// ---------------------------------------------------------------------------
// Implicit-GEMM conv3d via WMMA, tap-major K, TDM-staged B in LDS.
//   Block = 256 threads (8 waves); wave w owns 16x16 tile at M = blk*128+w*16.
//   All 8 waves share the B tile (32 k' x 16 n f16, double buffered in LDS,
//   DMA of chunk q+1 overlapped with WMMA of chunk q).
//   A: branch-free per-lane coalesced f32 global loads (address clamped to 0
//   for out-of-bounds slots, value zeroed by cndmask) so all 16 loads of a
//   chunk issue back-to-back with one loadcnt wait.
// ---------------------------------------------------------------------------
template<int KT>
__global__ __launch_bounds__(256)
void k_conv3d_wmma(const float* __restrict__ x, const _Float16* __restrict__ wP,
                   const float* __restrict__ bias, float* __restrict__ y,
                   int Cin, int Cout, int D, int H, int W, int Kp, int lcp)
{
    __shared__ _Float16 shB[2][16][32];

    const int tid    = threadIdx.x;
    const int lane   = tid & 31;
    const int wave   = tid >> 5;
    const int laneLo = lane & 15;
    const int laneHi = lane >> 4;
    const int HW = H * W;
    const long MV = (long)D * HW;
    const long mBase = (long)blockIdx.x * 128 + wave * 16;
    const int  nBase = blockIdx.y * 16;
    const int  Np    = gridDim.y * 16;
    const int  cmask = (1 << lcp) - 1;
    const int  PAD   = (KT == 27) ? 1 : 0;
    const int  nChunks = Kp >> 5;
    const int  tapPair = (lcp == 4) ? 1 : 0;   // two taps per 32-chunk (Cp==16)

    long m = mBase + laneLo;
    int oz = (int)(m / HW);
    int rr = (int)(m - (long)oz * HW);
    int oy = rr / W;
    int ox = rr - oy * W;
    int n  = nBase + laneLo;

    // prologue: stage chunk 0
    stage_b_tile(wP, nBase, 0, Kp, Np, &shB[0][0][0]);
    stage_wait();
    __syncthreads();

    v8f c = {};
    int cur = 0;
    for (int q = 0; q < nChunks; ++q) {
        const int k0p = q << 5;

        // kick DMA for next chunk (overlaps with this chunk's math)
        if (q + 1 < nChunks)
            stage_b_tile(wP, nBase, k0p + 32, Kp, Np, &shB[cur ^ 1][0][0]);

        // per-chunk tap predicates (1 tap, or 2 when Cp==16)
        const int tapBase = k0p >> lcp;
        long sbase0, sbase1;
        int  valid0, valid1;
#pragma unroll
        for (int t = 0; t < 2; ++t) {
            int tap = tapBase + t;
            int ok = ((t == 0) | tapPair) & (tap < KT ? 1 : 0);
            int kd = 0, kh = 0, kw = 0;
            if (KT == 27) {
                int tt = (tap < 27) ? tap : 0;
                kd = tt / 9; int r2 = tt - kd * 9;
                kh = r2 / 3; kw = r2 - kh * 3;
            }
            int iz = oz + kd - PAD, iy = oy + kh - PAD, ix = ox + kw - PAD;
            ok = ok && iz >= 0 && iz < D && iy >= 0 && iy < H && ix >= 0 && ix < W;
            long sb = ((long)iz * H + iy) * (long)W + ix;
            if (t == 0) { valid0 = ok; sbase0 = sb; }
            else        { valid1 = ok; sbase1 = sb; }
        }
        if (valid0 && (q + 1 < nChunks))
            __builtin_prefetch(x + sbase0, 0, 0);         // global_prefetch_b8

        // A fragment (16-bit A 16x32 lane layout) -- branch-free loads
        v16h a;
#pragma unroll
        for (int s = 0; s < 16; ++s) {
            int kl = s + laneHi * 16;
            int kk = k0p + kl;
            int t   = (kk >> lcp) - tapBase;
            int cin = kk & cmask;
            int ok  = (t ? valid1 : valid0) & (cin < Cin ? 1 : 0);
            long idx = (long)cin * MV + (t ? sbase1 : sbase0);
            float v = x[ok ? idx : 0];      // clamped address, unconditional load
            a[s] = (_Float16)(ok ? v : 0.f);
        }

        // B fragment: 32 contiguous bytes from LDS (lane's column, K half)
        v16h b = *(const v16h*)(&shB[cur][laneLo][laneHi * 16]);

        c = __builtin_amdgcn_wmma_f32_16x16x32_f16(
                false, a, false, b, (short)0, c, false, false);

        stage_wait();          // DMA for next chunk complete
        __syncthreads();       // everyone done reading shB[cur]
        cur ^= 1;
    }

    if (n < Cout) {
        float bv = bias[n];
#pragma unroll
        for (int r = 0; r < 8; ++r)
            y[(long)n * MV + mBase + r + laneHi * 8] = c[r] + bv;
    }
}

// ---------------------------------------------------------------------------
// BatchNorm (training-mode, weight=1 bias=0)
// ---------------------------------------------------------------------------
__global__ __launch_bounds__(256)
void k_bn_stats(const float* __restrict__ x, float* __restrict__ mean,
                float* __restrict__ var, long Mv)
{
    __shared__ float ss[256];
    __shared__ float s2[256];
    int c = blockIdx.x;
    const float* p = x + (long)c * Mv;
    float a = 0.f, b = 0.f;
    for (long i = threadIdx.x; i < Mv; i += 256) { float v = p[i]; a += v; b += v * v; }
    ss[threadIdx.x] = a; s2[threadIdx.x] = b;
    __syncthreads();
    for (int st = 128; st > 0; st >>= 1) {
        if ((int)threadIdx.x < st) {
            ss[threadIdx.x] += ss[threadIdx.x + st];
            s2[threadIdx.x] += s2[threadIdx.x + st];
        }
        __syncthreads();
    }
    if (threadIdx.x == 0) {
        float mu = ss[0] / (float)Mv;
        mean[c] = mu;
        var[c]  = s2[0] / (float)Mv - mu * mu;
    }
}

__global__ __launch_bounds__(256)
void k_bn_apply(const float* __restrict__ x, const float* __restrict__ mean,
                const float* __restrict__ var, float* __restrict__ y,
                long Mv, long total, int relu)
{
    long i = (long)blockIdx.x * 256 + threadIdx.x;
    if (i >= total) return;
    int c = (int)(i / Mv);
    float v = (x[i] - mean[c]) * rsqrtf(var[c] + 1e-5f);
    if (relu) v = fmaxf(v, 0.f);
    y[i] = v;
}

__global__ __launch_bounds__(256)
void k_add_relu(const float* __restrict__ a, const float* __restrict__ b,
                float* __restrict__ y, long n)
{
    long i = (long)blockIdx.x * 256 + threadIdx.x;
    if (i < n) y[i] = fmaxf(a[i] + b[i], 0.f);
}

__global__ __launch_bounds__(256)
void k_maxpool2(const float* __restrict__ x, float* __restrict__ y,
                int C, int Di, int Hi, int Wi)
{
    int Do = Di >> 1, Ho = Hi >> 1, Wo = Wi >> 1;
    long Mo = (long)Do * Ho * Wo;
    long total = (long)C * Mo;
    long i = (long)blockIdx.x * 256 + threadIdx.x;
    if (i >= total) return;
    int c = (int)(i / Mo);
    long r = i - (long)c * Mo;
    int z = (int)(r / ((long)Ho * Wo));
    int r2 = (int)(r - (long)z * Ho * Wo);
    int yy = r2 / Wo;
    int xx = r2 - yy * Wo;
    float mx = -3.402823466e38f;
    for (int dz = 0; dz < 2; ++dz)
        for (int dy = 0; dy < 2; ++dy)
            for (int dx = 0; dx < 2; ++dx) {
                float v = x[(((long)c * Di + (z * 2 + dz)) * Hi + (yy * 2 + dy)) * Wi + (xx * 2 + dx)];
                mx = fmaxf(mx, v);
            }
    y[i] = mx;
}

// ConvTranspose3d, kernel=2 stride=2. w layout (Cin, Cout, 2,2,2).
__global__ __launch_bounds__(256)
void k_deconv_k2(const float* __restrict__ x, const float* __restrict__ w,
                 const float* __restrict__ bias, float* __restrict__ y,
                 int Cin, int Cout, int Di, int Hi, int Wi)
{
    int Do = Di * 2, Ho = Hi * 2, Wo = Wi * 2;
    long Mo = (long)Do * Ho * Wo;
    long total = (long)Cout * Mo;
    long i = (long)blockIdx.x * 256 + threadIdx.x;
    if (i >= total) return;
    int co = (int)(i / Mo);
    long r = i - (long)co * Mo;
    int zo = (int)(r / ((long)Ho * Wo));
    int r2 = (int)(r - (long)zo * Ho * Wo);
    int yo = r2 / Wo;
    int xo = r2 - yo * Wo;
    int z = zo >> 1, dz = zo & 1, yy = yo >> 1, dy = yo & 1, xx = xo >> 1, dx = xo & 1;
    long Mi = (long)Di * Hi * Wi;
    long pin = ((long)z * Hi + yy) * Wi + xx;
    int woff = (dz * 2 + dy) * 2 + dx;
    float acc = bias[co];
    for (int ci = 0; ci < Cin; ++ci)
        acc += x[(long)ci * Mi + pin] * w[((long)ci * Cout + co) * 8 + woff];
    y[i] = acc;
}

// TCE: channel-affinity A[i,j] = adj[i,j] * g(cadj[j]-cadj[i])  (g symmetric)
__global__ __launch_bounds__(256)
void k_tce_s(const float* __restrict__ cadj, const float* __restrict__ adj,
             float* __restrict__ A)
{
    int idx = blockIdx.x * 256 + threadIdx.x;
    int i = idx >> 8, j = idx & 255;
    float d = cadj[j] - cadj[i];
    float sg = 1.f / (1.f + __expf(-d));
    float g = fabsf(fabsf(sg - 0.5f) - 0.5f) * 2.f;
    A[idx] = adj[idx] * g;
}

// att[i,n] = relu( (sum_j A[i,j]*fea[j,n]) * para[i,n] ),  fea = 256 x 384
__global__ __launch_bounds__(384)
void k_tce_mm(const float* __restrict__ A, const float* __restrict__ fea,
              const float* __restrict__ para, float* __restrict__ out)
{
    int i = blockIdx.x;
    int n = threadIdx.x;
    float acc = 0.f;
    for (int j = 0; j < 256; ++j)
        acc += A[i * 256 + j] * fea[j * 384 + n];
    acc *= para[i * 384 + n];
    out[i * 384 + n] = fmaxf(acc, 0.f);
}

// fuzzy membership: f = exp(-sum_c ((x-mu_c)/sigma_c)^2),  FUZZY_N=2, row 0
__global__ __launch_bounds__(256)
void k_fuzzy(const float* __restrict__ x, const float* __restrict__ mu,
             const float* __restrict__ sg, float* __restrict__ f, long n)
{
    long i = (long)blockIdx.x * 256 + threadIdx.x;
    if (i >= n) return;
    float v = x[i];
    float d0 = (v - mu[0]) / sg[0];
    float d1 = (v - mu[1]) / sg[1];
    f[i] = __expf(-(d0 * d0 + d1 * d1));
}

__global__ __launch_bounds__(256)
void k_sigmoid(const float* __restrict__ x, float* __restrict__ y, long n)
{
    long i = (long)blockIdx.x * 256 + threadIdx.x;
    if (i < n) y[i] = 1.f / (1.f + __expf(-x[i]));
}

// ---------------------------------------------------------------------------
extern "C" void kernel_launch(void* const* d_in, const int* in_sizes, int n_in,
                              void* d_out, int out_size, void* d_ws, size_t ws_size,
                              hipStream_t stream)
{
    (void)in_sizes; (void)n_in; (void)out_size; (void)ws_size;

    int p = 0;
    auto nxt = [&]() { return (const float*)d_in[p++]; };

    struct Res { const float *w1,*b1,*w2,*b2,*ws,*bs; };
    struct Dec { const float *w1,*b1,*w2,*b2; };
    struct Bce { const float *c1w,*c1b,*c2w,*c2b,*mu,*sigma; };
    auto getRes = [&]() { Res r; r.w1=nxt(); r.b1=nxt(); r.w2=nxt(); r.b2=nxt(); r.ws=nxt(); r.bs=nxt(); return r; };
    auto getDec = [&]() { Dec r; r.w1=nxt(); r.b1=nxt(); r.w2=nxt(); r.b2=nxt(); return r; };
    auto getBce = [&]() { Bce r; r.c1w=nxt(); r.c1b=nxt(); r.c2w=nxt(); r.c2b=nxt(); r.mu=nxt(); r.sigma=nxt(); return r; };

    // setup_inputs() insertion order: x, then make_params insertion order
    const float* x = nxt();
    Res enc0 = getRes(), enc1 = getRes(), enc2 = getRes(), enc3 = getRes(), enc4 = getRes();
    const float* tce_adj  = nxt();
    const float* tce_para = nxt();
    const float* dc4w = nxt(); const float* dc4b = nxt();
    const float* dc3w = nxt(); const float* dc3b = nxt();
    const float* dc2w = nxt(); const float* dc2b = nxt();
    const float* dc1w = nxt(); const float* dc1b = nxt();
    Bce f4 = getBce(), f3 = getBce(), f2 = getBce(), f1 = getBce();
    Dec de4 = getDec(), de3 = getDec(), de2 = getDec(), de1 = getDec();
    const float* finw = nxt(); const float* finb = nxt();

    // ---- workspace (bump allocator) ----
    const long V0 = 96L * 128 * 128, V1 = 48L * 64 * 64, V2 = 24L * 32 * 32,
               V3 = 12L * 16 * 16, V4 = 6L * 8 * 8;
    float* Wp = (float*)d_ws;
    size_t off = 0;
    auto alloc = [&](size_t nf) { float* r = Wp + off; off += nf; return r; };
    float* E0  = alloc(16 * V0);
    float* E1  = alloc(32 * V1);
    float* E2  = alloc(64 * V2);
    float* E3  = alloc(128 * V3);
    float* E4  = alloc(256 * V4);
    float* ATT = alloc(256 * V4);
    float* T1  = alloc(16 * V0);
    float* T2  = alloc(16 * V0);
    float* UP  = alloc(16 * V0);
    float* DB  = alloc(16 * V0);
    float* CAT = alloc(32 * V0);
    float* X1  = alloc(V0);
    float* FZ  = alloc(V0);
    float* SM  = alloc(256 * 256);
    float* MEAN = alloc(256);
    float* VAR  = alloc(256);
    float* CADJ = alloc(256);
    float* CVAR = alloc(256);
    _Float16* WPACK = (_Float16*)alloc(1 << 20);   // 2M halves, max need ~1.77M

    auto gr = [](long n) { return dim3((unsigned)((n + 255) / 256)); };

    // pack weights + run WMMA conv (tap-major K)
    auto conv = [&](int kt, const float* in, const float* w_, const float* b_, float* out,
                    int ci, int co, int D_, int H_, int W_) {
        int Cp = (ci >= 32) ? ci : 16;                  // Cin is 1,16,32,64,128,256
        int lcp = 0; while ((1 << lcp) < Cp) ++lcp;
        int Kp = ((kt * Cp + 31) / 32) * 32;
        int Np = ((co + 15) / 16) * 16;
        k_pack_w<<<gr((long)Np * Kp), dim3(256), 0, stream>>>(
            w_, WPACK, ci, co, kt, Cp, Kp, Np, lcp);
        dim3 g((unsigned)(((long)D_ * H_ * W_) / 128), (unsigned)(Np / 16));
        if (kt == 27)
            k_conv3d_wmma<27><<<g, dim3(256), 0, stream>>>(in, WPACK, b_, out, ci, co, D_, H_, W_, Kp, lcp);
        else
            k_conv3d_wmma<1 ><<<g, dim3(256), 0, stream>>>(in, WPACK, b_, out, ci, co, D_, H_, W_, Kp, lcp);
    };
    auto bn = [&](const float* in, float* out, int C, long Mv, int relu) {
        k_bn_stats<<<dim3((unsigned)C), dim3(256), 0, stream>>>(in, MEAN, VAR, Mv);
        long tot = (long)C * Mv;
        k_bn_apply<<<gr(tot), dim3(256), 0, stream>>>(in, MEAN, VAR, out, Mv, tot, relu);
    };
    auto resenc = [&](const float* in, const Res& R, int ci, int co,
                      int D_, int H_, int W_, float* out) {
        long Mv = (long)D_ * H_ * W_;
        conv(1,  in, R.ws, R.bs, UP, ci, co, D_, H_, W_);        // residual shortcut
        conv(27, in, R.w1, R.b1, T1, ci, co, D_, H_, W_);
        bn(T1, T2, co, Mv, 1);
        conv(27, T2, R.w2, R.b2, T1, co, co, D_, H_, W_);
        bn(T1, T2, co, Mv, 1);
        k_add_relu<<<gr((long)co * Mv), dim3(256), 0, stream>>>(T2, UP, out, (long)co * Mv);
    };

    // ---------------- encoder ----------------
    resenc(x,  enc0, 1,  16, 96, 128, 128, E0);
    k_maxpool2<<<gr(16L * V1), dim3(256), 0, stream>>>(E0, DB, 16, 96, 128, 128);
    resenc(DB, enc1, 16, 32, 48, 64, 64, E1);
    k_maxpool2<<<gr(32L * V2), dim3(256), 0, stream>>>(E1, DB, 32, 48, 64, 64);
    resenc(DB, enc2, 32, 64, 24, 32, 32, E2);
    k_maxpool2<<<gr(64L * V3), dim3(256), 0, stream>>>(E2, DB, 64, 24, 32, 32);
    resenc(DB, enc3, 64, 128, 12, 16, 16, E3);
    k_maxpool2<<<gr(128L * V4), dim3(256), 0, stream>>>(E3, DB, 128, 12, 16, 16);
    resenc(DB, enc4, 128, 256, 6, 8, 8, E4);

    // ---------------- TCE attention ----------------
    k_bn_stats<<<dim3(256), dim3(256), 0, stream>>>(E4, CADJ, CVAR, V4);  // channel means
    k_tce_s<<<dim3(256), dim3(256), 0, stream>>>(CADJ, tce_adj, SM);
    k_tce_mm<<<dim3(256), dim3(384), 0, stream>>>(SM, E4, tce_para, ATT);

    // ---------------- decoder ----------------
    const float* dws[4]   = { dc4w, dc3w, dc2w, dc1w };
    const float* dbs[4]   = { dc4b, dc3b, dc2b, dc1b };
    Bce bcs[4]            = { f4, f3, f2, f1 };
    Dec dcs[4]            = { de4, de3, de2, de1 };
    const float* skips[4] = { E3, E2, E1, E0 };

    const float* prev = ATT;
    int pC = 256, pD = 6, pH = 8, pW = 8;
    for (int L = 0; L < 4; ++L) {
        int oD = pD * 2, oH = pH * 2, oW = pW * 2;
        long Mv = (long)oD * oH * oW;
        int upC = pC / 2;          // 128, 64, 32, 16
        int ch  = upC * 2;         // bce fusion channels

        k_deconv_k2<<<gr((long)upC * Mv), dim3(256), 0, stream>>>(
            prev, dws[L], dbs[L], UP, pC, upC, pD, pH, pW);

        hipMemcpyAsync(CAT, skips[L], (size_t)upC * Mv * sizeof(float),
                       hipMemcpyDeviceToDevice, stream);
        hipMemcpyAsync(CAT + (size_t)upC * Mv, UP, (size_t)upC * Mv * sizeof(float),
                       hipMemcpyDeviceToDevice, stream);

        // BCE: c1 (ch->1), fuzzy, BN, c2 (1->ch), BN
        conv(27, CAT, bcs[L].c1w, bcs[L].c1b, X1, ch, 1, oD, oH, oW);
        k_fuzzy<<<gr(Mv), dim3(256), 0, stream>>>(X1, bcs[L].mu, bcs[L].sigma, FZ, Mv);
        bn(FZ, X1, 1, Mv, 0);
        conv(27, X1, bcs[L].c2w, bcs[L].c2b, T1, 1, ch, oD, oH, oW);
        bn(T1, T2, ch, Mv, 0);

        // decoder block: ch -> ch/2
        int oc = ch / 2;
        conv(27, T2, dcs[L].w1, dcs[L].b1, T1, ch, oc, oD, oH, oW);
        bn(T1, T2, oc, Mv, 1);
        conv(27, T2, dcs[L].w2, dcs[L].b2, T1, oc, oc, oD, oH, oW);
        bn(T1, DB, oc, Mv, 1);

        prev = DB; pC = oc; pD = oD; pH = oH; pW = oW;
    }

    // ---------------- head ----------------
    conv(1, DB, finw, finb, T1, 16, 2, 96, 128, 128);
    k_sigmoid<<<gr(2 * V0), dim3(256), 0, stream>>>(T1, (float*)d_out, 2 * V0);
}
